// VectorQuantization_87110526698166
// MI455X (gfx1250) — compile-verified, matrix operand order checked by reading, not run on hardware
//
#include <hip/hip_runtime.h>
#include <hip/hip_bf16.h>
#include <stdint.h>

#define B_ROWS 32768
#define DIM    512
#define KCODES 8192
#define TILE_N 32          // codes per TDM tile

typedef __attribute__((ext_vector_type(16))) __bf16 v16bf;
typedef __attribute__((ext_vector_type(8)))  float  v8f;
typedef __attribute__((ext_vector_type(4)))  unsigned int u32x4;
typedef __attribute__((ext_vector_type(8)))  int    i32x8;
typedef __attribute__((ext_vector_type(4)))  int    i32x4;
typedef __attribute__((ext_vector_type(4)))  unsigned int v4u;

union FragBF { v4u u[2]; v16bf v; };

__device__ __forceinline__ unsigned short f2bf(float f) {
    unsigned u = __float_as_uint(f);
    unsigned r = u + 0x7FFFu + ((u >> 16) & 1u);   // round-to-nearest-even
    return (unsigned short)(r >> 16);
}

// ---------------------------------------------------------------------------
// Kernel A: convert x (f32 -> bf16), zero loss accumulator
// ---------------------------------------------------------------------------
__global__ __launch_bounds__(256) void convert_x_kernel(
    const float* __restrict__ x, unsigned short* __restrict__ xh,
    float* __restrict__ accum, int n)
{
    int i = blockIdx.x * blockDim.x + threadIdx.x;
    if (i == 0) *accum = 0.0f;
    for (; i < n; i += gridDim.x * blockDim.x) xh[i] = f2bf(x[i]);
}

// ---------------------------------------------------------------------------
// Kernel B: per-code ||w||^2 (one wave per code row) + convert w -> bf16
// ---------------------------------------------------------------------------
__global__ __launch_bounds__(256) void prep_w_kernel(
    const float* __restrict__ w, unsigned short* __restrict__ wh,
    float* __restrict__ wnorm)
{
    int wave = (blockIdx.x * blockDim.x + threadIdx.x) >> 5;
    int lane = threadIdx.x & 31;
    if (wave >= KCODES) return;
    const float* row = w + (size_t)wave * DIM;
    unsigned short* orow = wh + (size_t)wave * DIM;
    float s = 0.0f;
    for (int j = lane; j < DIM; j += 32) {
        float v = row[j];
        s += v * v;
        orow[j] = f2bf(v);
    }
    #pragma unroll
    for (int m = 16; m; m >>= 1) s += __shfl_xor(s, m, 32);
    if (lane == 0) wnorm[wave] = s;
}

// ---------------------------------------------------------------------------
// TDM: load a TILE_N-code x 512-elem bf16 tile (32 KB) from global into LDS.
// D# built per CDNA5 ISA (2D tile, data_size=2B).
// ---------------------------------------------------------------------------
__device__ __forceinline__ void tdm_load_tile(unsigned lds_off,
                                              const unsigned short* gptr)
{
    unsigned long long ga = (unsigned long long)(uintptr_t)gptr;
    u32x4 g0; i32x8 g1; i32x4 g2; i32x4 g3; i32x8 g4;
    // group0: count=1, lds_addr, global_addr[56:0], type=2 ("image")
    g0[0] = 1u;
    g0[1] = lds_off;
    g0[2] = (unsigned)(ga & 0xFFFFFFFFu);
    g0[3] = (unsigned)((ga >> 32) & 0x1FFFFFFu) | (2u << 30);
    // group1: data_size=2B; tensor_dim0=512, tensor_dim1=8192;
    //         tile_dim0=512, tile_dim1=TILE_N; tensor_dim0_stride=512
    g1[0] = (int)(1u << 16);          // wg_mask=0, data_size=1 (2 bytes)
    g1[1] = (int)(512u << 16);        // tensor_dim0[15:0] in bits [63:48]
    g1[2] = (int)(8192u << 16);       // dim0 hi=0 | tensor_dim1[15:0] in [95:80]
    g1[3] = (int)(512u << 16);        // dim1 hi=0 | tile_dim0 in [127:112]
    g1[4] = TILE_N;                   // tile_dim1, tile_dim2=0
    g1[5] = 512;                      // tensor_dim0_stride low 32
    g1[6] = 0;
    g1[7] = 0;
    g2[0] = g2[1] = g2[2] = g2[3] = 0;
    g3[0] = g3[1] = g3[2] = g3[3] = 0;
    g4[0] = g4[1] = g4[2] = g4[3] = 0;
    g4[4] = g4[5] = g4[6] = g4[7] = 0;
    __builtin_amdgcn_tensor_load_to_lds(g0, g1, g2, g3, g4, 0);
}

// ---------------------------------------------------------------------------
// Kernel C: fused GEMM + argmin.
// Block = 128 threads = 4 waves; each wave owns TWO 16-row x-tiles (32 rows,
// A-fragments in registers) so every B fragment read from LDS feeds 2 WMMAs
// (halves LDS traffic per WMMA). Weight tiles (32 codes) double-buffered in
// LDS via TDM. score = ||w||^2 - 2*(x.w); per-lane running min, then
// cross-lane argmin reduction within 16-lane groups.
// ---------------------------------------------------------------------------
__global__ __launch_bounds__(128) void vq_argmin_kernel(
    const unsigned short* __restrict__ xh,
    const unsigned short* __restrict__ wh,
    const float* __restrict__ wnorm,
    int* __restrict__ ind)
{
    __shared__ __align__(128) unsigned char lds[2][TILE_N * DIM * 2]; // 2x32KB

    const int wave = threadIdx.x >> 5;
    const int lane = threadIdx.x & 31;
    const int n16  = lane & 15;
    const int koff = (lane & 16) ? 8 : 0;     // k-base for this lane half

    const int rowBase = blockIdx.x * 128 + wave * 32;

    // --- preload A fragments: two 16-row tiles, all of d=512 ---
    FragBF afrag[2][16];
    #pragma unroll
    for (int t = 0; t < 2; ++t) {
        const unsigned short* arow = xh + (size_t)(rowBase + t * 16 + n16) * DIM;
        #pragma unroll
        for (int kt = 0; kt < 16; ++kt) {
            afrag[t][kt].u[0] = *(const v4u*)(arow + kt * 32 + koff);
            afrag[t][kt].u[1] = *(const v4u*)(arow + kt * 32 + koff + 16);
        }
    }

    float minv[2][8];
    int   mini[2][8];
    #pragma unroll
    for (int t = 0; t < 2; ++t)
        #pragma unroll
        for (int r = 0; r < 8; ++r) { minv[t][r] = 3.402823466e38f; mini[t][r] = 0; }

    const unsigned lds0 = (unsigned)(uintptr_t)&lds[0][0];
    const unsigned lds1 = (unsigned)(uintptr_t)&lds[1][0];
    const int NTILES = KCODES / TILE_N;   // 256

    if (wave == 0) tdm_load_tile(lds0, wh);   // prologue: tile 0 -> buf0

    for (int nt = 0; nt < NTILES; ++nt) {
        if (wave == 0) __builtin_amdgcn_s_wait_tensorcnt(0); // tile nt landed
        __syncthreads();                                      // visible to all
        if (wave == 0 && (nt + 1) < NTILES)                   // prefetch nt+1
            tdm_load_tile((nt & 1) ? lds0 : lds1,
                          wh + (size_t)(nt + 1) * TILE_N * DIM);

        const unsigned short* buf =
            (const unsigned short*)((nt & 1) ? lds[1] : lds[0]);

        #pragma unroll
        for (int c = 0; c < 2; ++c) {                 // two 16-code subtiles
            const unsigned short* bbase = buf + (c * 16 + n16) * DIM;

            v8f acc0 = {0.f,0.f,0.f,0.f,0.f,0.f,0.f,0.f};
            v8f acc1 = {0.f,0.f,0.f,0.f,0.f,0.f,0.f,0.f};
            #pragma unroll
            for (int kt = 0; kt < 16; ++kt) {
                FragBF bf;
                bf.u[0] = *(const v4u*)(bbase + kt * 32 + koff);
                bf.u[1] = *(const v4u*)(bbase + kt * 32 + koff + 16);
                acc0 = __builtin_amdgcn_wmma_f32_16x16x32_bf16(
                    false, afrag[0][kt].v, false, bf.v, (short)0, acc0, false, false);
                acc1 = __builtin_amdgcn_wmma_f32_16x16x32_bf16(
                    false, afrag[1][kt].v, false, bf.v, (short)0, acc1, false, false);
            }

            const int   ngl = nt * TILE_N + c * 16 + n16;
            const float wn  = wnorm[ngl];
            #pragma unroll
            for (int r = 0; r < 8; ++r) {
                float s0 = wn - 2.0f * acc0[r];
                if (s0 < minv[0][r]) { minv[0][r] = s0; mini[0][r] = ngl; }
                float s1 = wn - 2.0f * acc1[r];
                if (s1 < minv[1][r]) { minv[1][r] = s1; mini[1][r] = ngl; }
            }
        }
    }

    // --- cross-lane (16-lane group) argmin reduction, both row tiles ---
    #pragma unroll
    for (int mask = 1; mask < 16; mask <<= 1) {
        #pragma unroll
        for (int t = 0; t < 2; ++t) {
            #pragma unroll
            for (int r = 0; r < 8; ++r) {
                float ov = __shfl_xor(minv[t][r], mask, 32);
                int   oi = __shfl_xor(mini[t][r], mask, 32);
                if (ov < minv[t][r] || (ov == minv[t][r] && oi < mini[t][r])) {
                    minv[t][r] = ov; mini[t][r] = oi;
                }
            }
        }
    }
    if (n16 == 0) {
        #pragma unroll
        for (int t = 0; t < 2; ++t) {
            int rbase = rowBase + t * 16 + ((lane >> 4) * 8);
            #pragma unroll
            for (int r = 0; r < 8; ++r) ind[rbase + r] = mini[t][r];
        }
    }
}

// ---------------------------------------------------------------------------
// Kernel D: quantized = weight[ind] (f32), accumulate sum((x-q)^2)
// ---------------------------------------------------------------------------
__global__ __launch_bounds__(256) void gather_loss_kernel(
    const float* __restrict__ x, const float* __restrict__ w,
    const int* __restrict__ ind, float* __restrict__ outq,
    float* __restrict__ accum)
{
    int b = blockIdx.x;
    int q = ind[b];
    const float* xr   = x + (size_t)b * DIM;
    const float* wr   = w + (size_t)q * DIM;
    float*       orow = outq + (size_t)b * DIM;
    float s = 0.0f;
    for (int j = threadIdx.x; j < DIM; j += 256) {
        float wv = wr[j];
        orow[j] = wv;                 // quantized_st == quantized numerically
        float d = xr[j] - wv;
        s += d * d;
    }
    #pragma unroll
    for (int m = 16; m; m >>= 1) s += __shfl_xor(s, m, 32);
    __shared__ float red[8];
    if ((threadIdx.x & 31) == 0) red[threadIdx.x >> 5] = s;
    __syncthreads();
    if (threadIdx.x == 0) {
        float t = 0.0f;
        #pragma unroll
        for (int i = 0; i < 8; ++i) t += red[i];
        atomicAdd(accum, t);
    }
}

// ---------------------------------------------------------------------------
// Kernel E: loss = (BETA + 1) * mean = 1.25 * sum / (B*d)
// ---------------------------------------------------------------------------
__global__ void finalize_kernel(const float* __restrict__ accum,
                                float* __restrict__ lossOut)
{
    *lossOut = 1.25f * (*accum) / (float)((long long)B_ROWS * DIM);
}

// ---------------------------------------------------------------------------
extern "C" void kernel_launch(void* const* d_in, const int* in_sizes, int n_in,
                              void* d_out, int out_size, void* d_ws, size_t ws_size,
                              hipStream_t stream)
{
    const float* x = (const float*)d_in[0];   // [B, d] f32
    const float* w = (const float*)d_in[1];   // [K, d] f32

    float* outq  = (float*)d_out;                                 // [B*d]
    int*   ind   = (int*)((float*)d_out + (size_t)B_ROWS * DIM);  // [B] int32
    float* lossp = (float*)d_out + (size_t)B_ROWS * DIM + B_ROWS; // scalar

    // workspace layout (16B-aligned sections)
    float*          accum = (float*)d_ws;
    float*          wnorm = (float*)((char*)d_ws + 256);
    unsigned short* wh    = (unsigned short*)((char*)d_ws + 256 + 32768);
    unsigned short* xh    = (unsigned short*)((char*)d_ws + 256 + 32768 +
                                              (size_t)KCODES * DIM * 2);

    convert_x_kernel<<<2048, 256, 0, stream>>>(x, xh, accum, B_ROWS * DIM);
    prep_w_kernel<<<KCODES / 8, 256, 0, stream>>>(w, wh, wnorm);
    vq_argmin_kernel<<<B_ROWS / 128, 128, 0, stream>>>(xh, wh, wnorm, ind);
    gather_loss_kernel<<<B_ROWS, 256, 0, stream>>>(x, w, ind, outq, accum);
    finalize_kernel<<<1, 1, 0, stream>>>(accum, lossp);
}
